// Mamba2LM_257698037817
// MI455X (gfx1250) — compile-verified, matrix operand order checked by reading, not run on hardware
//
#include <hip/hip_runtime.h>
#include <hip/hip_bf16.h>

// ---------------- model constants ----------------
constexpr int SEQ     = 2048;
constexpr int DMODEL  = 768;
constexpr int NLAYER  = 16;
constexpr int DSTATE  = 64;
constexpr int DINNER  = 1536;
constexpr int NHEADS  = 24;
constexpr int HEAD    = 64;
constexpr int CONVDIM = 1664;   // DINNER + 2*DSTATE
constexpr int DINPROJ = 3224;   // 2*DINNER + 2*DSTATE + NHEADS
constexpr int CHUNKT  = 256;
constexpr int NCHUNK  = 8;
constexpr int VOCAB   = 32000;
constexpr int NPAD_IP = 3328;   // DINPROJ padded to multiple of 128

typedef __attribute__((ext_vector_type(16))) __bf16 bf16x16;
typedef __attribute__((ext_vector_type(8)))  __bf16 bf16x8;
typedef __attribute__((ext_vector_type(8)))  float  f32x8;

__device__ __forceinline__ f32x8 wmma_bf16(bf16x16 a, bf16x16 b, f32x8 c) {
  return __builtin_amdgcn_wmma_f32_16x16x32_bf16(false, a, false, b, (short)0, c,
                                                 false, false);
}

// Async global->LDS 16-byte copy (per lane). LDS offset = low 32 bits of the
// generic shared-pointer address (aperture rule, ISA 10.2). Tracked by ASYNCcnt.
__device__ __forceinline__ void async_b128(const __bf16* lds_dst, const __bf16* g) {
  unsigned lds_off = (unsigned)(unsigned long long)lds_dst;
  unsigned long long ga = (unsigned long long)g;
  asm volatile("global_load_async_to_lds_b128 %0, %1, off"
               :: "v"(lds_off), "v"(ga)
               : "memory");
}

__device__ __forceinline__ void wait_async0() {
#if __has_builtin(__builtin_amdgcn_s_wait_asynccnt)
  __builtin_amdgcn_s_wait_asynccnt(0);
#else
  asm volatile("s_wait_asynccnt 0x0" ::: "memory");
#endif
}

// 16x32 bf16 fragment, contiguous K, NO bounds checks: two 16-byte loads.
// ISA 7.12.2 layout: lane&15 = row; lane<16 holds K {0..7,16..23}, lane>=16
// holds K {8..15,24..31}.  Requires ld % 8 == 0, k0 % 8 == 0, base 16B aligned.
__device__ __forceinline__ bf16x16 ldfrag_v(const __bf16* base, int ld,
                                            int i0, int k0, int lane) {
  const __bf16* p = base + (long)(i0 + (lane & 15)) * ld + k0 + (lane >> 4) * 8;
  bf16x8 lo = *(const bf16x8*)p;
  bf16x8 hi = *(const bf16x8*)(p + 16);
  return __builtin_shufflevector(lo, hi, 0, 1, 2, 3, 4, 5, 6, 7,
                                 8, 9, 10, 11, 12, 13, 14, 15);
}

// ---------------- elementwise / staging kernels ----------------

__global__ void embed_kernel(const int* ids, const float* emb, float* x) {
  int t = blockIdx.x;
  long row = (long)ids[t] * DMODEL;
  for (int i = threadIdx.x; i < DMODEL; i += 256)
    x[(long)t * DMODEL + i] = emb[row + i];
}

__global__ void rmsnorm_bf16_kernel(const float* x, const float* w, __bf16* out,
                                    int D, float eps) {
  int row = blockIdx.x;
  const float* xr = x + (long)row * D;
  __shared__ float red[256];
  float s = 0.0f;
  for (int i = threadIdx.x; i < D; i += 256) { float v = xr[i]; s += v * v; }
  red[threadIdx.x] = s;
  __syncthreads();
  for (int st = 128; st > 0; st >>= 1) {
    if (threadIdx.x < st) red[threadIdx.x] += red[threadIdx.x + st];
    __syncthreads();
  }
  float inv = rsqrtf(red[0] / (float)D + eps);
  for (int i = threadIdx.x; i < D; i += 256)
    out[(long)row * D + i] = (__bf16)(xr[i] * inv * w[i]);
}

// weight f32 [N,K] -> bf16 [Npad,K], zero-padding rows N..Npad
__global__ void cvt_w_kernel(const float* src, __bf16* dst, int N, int Npad, int K) {
  long gid = (long)blockIdx.x * 256 + threadIdx.x;
  long total = (long)Npad * K;
  if (gid >= total) return;
  int n = (int)(gid / K);
  int k = (int)(gid % K);
  dst[gid] = (__bf16)((n < N) ? src[(long)n * K + k] : 0.0f);
}

__global__ void conv_silu_kernel(const float* zx, const float* cw, const float* cb,
                                 float* xc) {
  long gid = (long)blockIdx.x * 256 + threadIdx.x;
  if (gid >= (long)SEQ * CONVDIM) return;
  int t  = (int)(gid / CONVDIM);
  int ch = (int)(gid % CONVDIM);
  float acc = cb[ch];
#pragma unroll
  for (int j = 0; j < 4; ++j) {
    int tt = t - 3 + j;
    if (tt >= 0) acc += zx[(long)tt * DINPROJ + DINNER + ch] * cw[ch * 4 + j];
  }
  xc[(long)t * CONVDIM + ch] = acc / (1.0f + expf(-acc));
}

__global__ void dt_kernel(const float* zx, const float* dtb, const float* alog,
                          float* dtv, float* dAv) {
  int gid = blockIdx.x * 256 + threadIdx.x;
  if (gid >= SEQ * NHEADS) return;
  int t = gid / NHEADS, h = gid % NHEADS;
  float raw = zx[(long)t * DINPROJ + (DINPROJ - NHEADS) + h] + dtb[h];
  float dt = (raw > 20.0f) ? raw : log1pf(expf(raw));
  dtv[gid] = dt;
  dAv[gid] = dt * (-expf(alog[h]));
}

__global__ void cumsum_kernel(const float* dAv, float* dAcs, float* atot) {
  int gid = blockIdx.x * 256 + threadIdx.x;
  if (gid >= NCHUNK * NHEADS) return;
  int c = gid / NHEADS, h = gid % NHEADS;
  float s = 0.0f;
  for (int i = 0; i < CHUNKT; ++i) {
    s += dAv[(long)(c * CHUNKT + i) * NHEADS + h];
    dAcs[((long)c * NHEADS + h) * CHUNKT + i] = s;
  }
  atot[gid] = s;
}

__global__ void stage_bc_kernel(const float* xc, __bf16* Bbf, __bf16* Cbf,
                                __bf16* BbfT) {
  int gid = blockIdx.x * 256 + threadIdx.x;
  if (gid >= SEQ * DSTATE) return;
  int t = gid / DSTATE, n = gid % DSTATE;
  float bv = xc[(long)t * CONVDIM + DINNER + n];
  float cv = xc[(long)t * CONVDIM + DINNER + DSTATE + n];
  Bbf[gid] = (__bf16)bv;
  BbfT[(long)n * SEQ + t] = (__bf16)bv;
  Cbf[gid] = (__bf16)cv;
}

__global__ void stage_x_kernel(const float* xc, const float* dtv, const float* dAcs,
                               const float* atot, __bf16* xdtT, __bf16* xddT) {
  long gid = (long)blockIdx.x * 256 + threadIdx.x;
  if (gid >= (long)SEQ * DINNER) return;
  int t = (int)(gid / DINNER);
  int i = (int)(gid % DINNER);
  int h = i / HEAD;
  int c = t / CHUNKT, r = t % CHUNKT;
  float xv = xc[(long)t * CONVDIM + i] * dtv[t * NHEADS + h];
  float cs = dAcs[((long)c * NHEADS + h) * CHUNKT + r];
  float dec = expf(atot[c * NHEADS + h] - cs);
  xdtT[(long)i * SEQ + t] = (__bf16)xv;
  xddT[(long)i * SEQ + t] = (__bf16)(dec * xv);
}

__global__ void gated_norm_kernel(const float* yb, const float* zx, const float* gw,
                                  __bf16* yn) {
  int t = blockIdx.x;
  __shared__ float red[256];
  float s = 0.0f;
  for (int i = threadIdx.x; i < DINNER; i += 256) {
    float z = zx[(long)t * DINPROJ + i];
    float u = yb[(long)t * DINNER + i] * (z / (1.0f + expf(-z)));
    s += u * u;
  }
  red[threadIdx.x] = s;
  __syncthreads();
  for (int st = 128; st > 0; st >>= 1) {
    if (threadIdx.x < st) red[threadIdx.x] += red[threadIdx.x + st];
    __syncthreads();
  }
  float inv = rsqrtf(red[0] / (float)DINNER + 1e-5f);
  for (int i = threadIdx.x; i < DINNER; i += 256) {
    float z = zx[(long)t * DINPROJ + i];
    float u = yb[(long)t * DINNER + i] * (z / (1.0f + expf(-z)));
    yn[(long)t * DINNER + i] = (__bf16)(u * inv * gw[i]);
  }
}

// -------- WMMA GEMM: C[M,N] = A[M,K](bf16) * W[Npad,K](bf16)^T (+residual) --------
// Block tile 64x128, 8 waves (2x4), each wave 2x2 WMMA tiles.  A/W tiles are
// staged to LDS via async global->LDS copies (ASYNCcnt), double-buffered, so
// each global byte is fetched exactly once per block; fragments come from LDS.
__global__ void gemm_nt_bf16_wmma(const __bf16* __restrict__ A,
                                  const __bf16* __restrict__ W,
                                  float* __restrict__ C,
                                  const float* __restrict__ residual,
                                  int M, int N, int K) {
  __shared__ __bf16 As[2][64 * 32];    // 4 KB x2
  __shared__ __bf16 Ws[2][128 * 32];   // 8 KB x2
  int tid = threadIdx.x;
  int lane = tid & 31, wave = tid >> 5;
  int wr = wave >> 2, wc = wave & 3;
  int rowblk = blockIdx.x * 64;
  int colblk = blockIdx.y * 128;

  auto stage = [&](int buf, int k0) {
    {
      int r = tid >> 2, col = (tid & 3) * 8;      // 64 rows x 4 chunks
      async_b128(&As[buf][r * 32 + col], A + (long)(rowblk + r) * K + k0 + col);
    }
#pragma unroll
    for (int q = 0; q < 2; ++q) {
      int idx = tid + q * 256;                    // 128 rows x 4 chunks
      int r = idx >> 2, col = (idx & 3) * 8;
      async_b128(&Ws[buf][r * 32 + col], W + (long)(colblk + r) * K + k0 + col);
    }
  };

  f32x8 acc00 = {}, acc01 = {}, acc10 = {}, acc11 = {};
  int KT = K >> 5;
  int buf = 0;
  stage(0, 0);
  for (int kt = 0; kt < KT; ++kt) {
    wait_async0();
    __syncthreads();                     // tile kt resident in LDS
    if (kt + 1 < KT) stage(buf ^ 1, (kt + 1) * 32);
    bf16x16 a0 = ldfrag_v(As[buf], 32, wr * 32, 0, lane);
    bf16x16 a1 = ldfrag_v(As[buf], 32, wr * 32 + 16, 0, lane);
    bf16x16 b0 = ldfrag_v(Ws[buf], 32, wc * 32, 0, lane);
    bf16x16 b1 = ldfrag_v(Ws[buf], 32, wc * 32 + 16, 0, lane);
    acc00 = wmma_bf16(a0, b0, acc00);
    acc01 = wmma_bf16(a0, b1, acc01);
    acc10 = wmma_bf16(a1, b0, acc10);
    acc11 = wmma_bf16(a1, b1, acc11);
    __syncthreads();                     // all waves done reading buf
    buf ^= 1;
  }

  int row0 = rowblk + wr * 32;
  int col0 = colblk + wc * 32;
  int nl = lane & 15, mo = (lane >> 4) * 8;
#pragma unroll
  for (int mt = 0; mt < 2; ++mt)
#pragma unroll
    for (int nt = 0; nt < 2; ++nt) {
      f32x8 acc = (mt == 0) ? (nt == 0 ? acc00 : acc01) : (nt == 0 ? acc10 : acc11);
      int n = col0 + nt * 16 + nl;
      if (n < N) {
#pragma unroll
        for (int r = 0; r < 8; ++r) {
          int m = row0 + mt * 16 + r + mo;
          float v = acc[r];
          if (residual) v += residual[(long)m * N + n];
          C[(long)m * N + n] = v;
        }
      }
    }
}

// ---------------- SSD intra-chunk: Y_diag + chunk states ----------------
__global__ void ssd_chunk_kernel(const __bf16* __restrict__ Bbf,
                                 const __bf16* __restrict__ Cbf,
                                 const __bf16* __restrict__ BbfT,
                                 const __bf16* __restrict__ xdtT,
                                 const __bf16* __restrict__ xddT,
                                 const float* __restrict__ dAcs_g,
                                 float* __restrict__ ybuf,
                                 float* __restrict__ stbuf) {
  int c = blockIdx.x, h = blockIdx.y;
  int tid = threadIdx.x, lane = tid & 31, wave = tid >> 5;
  int t0 = c * CHUNKT;

  __shared__ float  cs[CHUNKT];
  __shared__ __bf16 Gs[CHUNKT * 64];   // 32 KB slab: 256 rows x 64 s-columns

  for (int i = tid; i < CHUNKT; i += 256)
    cs[i] = dAcs_g[((long)c * NHEADS + h) * CHUNKT + i];
  __syncthreads();

  const __bf16* Cb = Cbf + (long)t0 * DSTATE;
  const __bf16* Bb = Bbf + (long)t0 * DSTATE;
  const __bf16* Xb = xdtT + (long)(h * HEAD) * SEQ + t0;   // [p][s] within chunk
  const __bf16* Xd = xddT + (long)(h * HEAD) * SEQ + t0;
  const __bf16* BT = BbfT + t0;                            // [n][s] within chunk

  f32x8 acc[2][4];
#pragma unroll
  for (int u = 0; u < 2; ++u)
#pragma unroll
    for (int v = 0; v < 4; ++v) acc[u][v] = (f32x8){};

  for (int sb = 0; sb < 4; ++sb) {
    int s0 = sb * 64;
    __syncthreads();
    // build masked-decay slab G[m, s0:s0+64]: 64 tiles, 8 per wave
    for (int q = 0; q < 8; ++q) {
      int tile = wave + 8 * q;
      int tm = tile >> 2, ts = tile & 3;
      int m0 = tm * 16, sg0 = s0 + ts * 16;
      f32x8 g = {};
      if (sg0 <= m0 + 15) {
        for (int k0 = 0; k0 < HEAD; k0 += 32) {
          bf16x16 a = ldfrag_v(Cb, DSTATE, m0, k0, lane);
          bf16x16 b = ldfrag_v(Bb, DSTATE, sg0, k0, lane);
          g = wmma_bf16(a, b, g);
        }
      }
      int n = lane & 15, mo = (lane >> 4) * 8;
#pragma unroll
      for (int r = 0; r < 8; ++r) {
        int m = m0 + r + mo;
        int s = sg0 + n;
        float v = (s <= m) ? g[r] * expf(cs[m] - cs[s]) : 0.0f;
        Gs[(long)m * 64 + (ts * 16 + n)] = (__bf16)v;
      }
    }
    __syncthreads();
    // Y_diag partial: acc += G[m, s0:s0+64] * X[s0:s0+64, p]
    for (int k0 = 0; k0 < 64; k0 += 32) {
      bf16x16 bx[4];
#pragma unroll
      for (int pt = 0; pt < 4; ++pt)
        bx[pt] = ldfrag_v(Xb, SEQ, pt * 16, s0 + k0, lane);
#pragma unroll
      for (int u = 0; u < 2; ++u) {
        bf16x16 a = ldfrag_v(Gs, 64, (wave + 8 * u) * 16, k0, lane);
#pragma unroll
        for (int pt = 0; pt < 4; ++pt)
          acc[u][pt] = wmma_bf16(a, bx[pt], acc[u][pt]);
      }
    }
  }

  // store Y_diag
  for (int u = 0; u < 2; ++u) {
    int m0 = (wave + 8 * u) * 16;
    int n = lane & 15, mo = (lane >> 4) * 8;
#pragma unroll
    for (int pt = 0; pt < 4; ++pt)
#pragma unroll
      for (int r = 0; r < 8; ++r) {
        int m = m0 + r + mo;
        ybuf[(long)(t0 + m) * DINNER + h * HEAD + pt * 16 + n] = acc[u][pt][r];
      }
  }

  // chunk states: st[p,n] = sum_s (decay X)[p,s] * B[s,n]
  for (int u = 0; u < 2; ++u) {
    int tt = wave + 8 * u;
    int p0 = (tt >> 2) * 16, n0 = (tt & 3) * 16;
    f32x8 sacc = {};
    for (int k0 = 0; k0 < CHUNKT; k0 += 32) {
      bf16x16 a = ldfrag_v(Xd, SEQ, p0, k0, lane);
      bf16x16 b = ldfrag_v(BT, SEQ, n0, k0, lane);
      sacc = wmma_bf16(a, b, sacc);
    }
    int n = lane & 15, mo = (lane >> 4) * 8;
#pragma unroll
    for (int r = 0; r < 8; ++r)
      stbuf[((long)c * NHEADS + h) * (HEAD * DSTATE) +
            (long)(p0 + r + mo) * DSTATE + n0 + n] = sacc[r];
  }
}

// inter-chunk recurrence; writes prev-states as bf16 [p][n]
__global__ void ssd_scan_kernel(const float* st, const float* atot, __bf16* pst) {
  int h = blockIdx.x, tid = threadIdx.x;
  float P[16];
#pragma unroll
  for (int j = 0; j < 16; ++j) P[j] = 0.0f;
  for (int c = 0; c < NCHUNK; ++c) {
    long base = ((long)c * NHEADS + h) * (HEAD * DSTATE);
    float e = expf(atot[c * NHEADS + h]);
#pragma unroll
    for (int j = 0; j < 16; ++j) {
      int idx = tid + j * 256;
      pst[base + idx] = (__bf16)P[j];
      P[j] = e * P[j] + st[base + idx];
    }
  }
}

// Y += exp(cs[l]) * (C · prev_state) + D_skip * xs
__global__ void ssd_off_kernel(const __bf16* __restrict__ Cbf,
                               const __bf16* __restrict__ pst,
                               const float* __restrict__ dAcs_g,
                               const float* __restrict__ xc,
                               const float* __restrict__ dskip,
                               float* __restrict__ ybuf) {
  int c = blockIdx.x, h = blockIdx.y;
  int tid = threadIdx.x, lane = tid & 31, wave = tid >> 5;
  int t0 = c * CHUNKT;
  __shared__ float cs[CHUNKT];
  for (int i = tid; i < CHUNKT; i += 256)
    cs[i] = dAcs_g[((long)c * NHEADS + h) * CHUNKT + i];
  __syncthreads();

  const __bf16* Cb = Cbf + (long)t0 * DSTATE;
  const __bf16* Pb = pst + ((long)c * NHEADS + h) * (HEAD * DSTATE);  // [p][n]
  float dsk = dskip[h];

  for (int u = 0; u < 2; ++u) {
    int m0 = (wave + 8 * u) * 16;
    f32x8 acc[4];
#pragma unroll
    for (int v = 0; v < 4; ++v) acc[v] = (f32x8){};
    for (int k0 = 0; k0 < DSTATE; k0 += 32) {
      bf16x16 a = ldfrag_v(Cb, DSTATE, m0, k0, lane);
#pragma unroll
      for (int pt = 0; pt < 4; ++pt) {
        bf16x16 b = ldfrag_v(Pb, DSTATE, pt * 16, k0, lane);
        acc[pt] = wmma_bf16(a, b, acc[pt]);
      }
    }
    int n = lane & 15, mo = (lane >> 4) * 8;
#pragma unroll
    for (int pt = 0; pt < 4; ++pt)
#pragma unroll
      for (int r = 0; r < 8; ++r) {
        int m = m0 + r + mo;
        int t = t0 + m;
        int p = pt * 16 + n;
        long yi = (long)t * DINNER + h * HEAD + p;
        float xsv = xc[(long)t * CONVDIM + h * HEAD + p];
        ybuf[yi] = ybuf[yi] + acc[pt][r] * expf(cs[m]) + dsk * xsv;
      }
  }
}

// ---------------- host orchestration ----------------

extern "C" void kernel_launch(void* const* d_in, const int* in_sizes, int n_in,
                              void* d_out, int out_size, void* d_ws, size_t ws_size,
                              hipStream_t stream) {
  const int*   ids  = (const int*)  d_in[0];
  const float* emb  = (const float*)d_in[1];
  const float* lnw  = (const float*)d_in[2];
  const float* ipw  = (const float*)d_in[3];
  const float* cwv  = (const float*)d_in[4];
  const float* cbv  = (const float*)d_in[5];
  const float* dtb  = (const float*)d_in[6];
  const float* alog = (const float*)d_in[7];
  const float* dsk  = (const float*)d_in[8];
  const float* gwv  = (const float*)d_in[9];
  const float* opw  = (const float*)d_in[10];
  const float* nfw  = (const float*)d_in[11];

  char* wsp = (char*)d_ws;
  auto carve = [&](size_t bytes) -> void* {
    void* p = (void*)wsp;
    wsp += (bytes + 255) & ~(size_t)255;
    return p;
  };
  float*  x0    = (float*) carve((size_t)SEQ * DMODEL * 4);
  float*  x1    = (float*) carve((size_t)SEQ * DMODEL * 4);
  __bf16* hnbf  = (__bf16*)carve((size_t)SEQ * DMODEL * 2);
  float*  zx    = (float*) carve((size_t)SEQ * DINPROJ * 4);
  float*  xc    = (float*) carve((size_t)SEQ * CONVDIM * 4);
  float*  dtv   = (float*) carve((size_t)SEQ * NHEADS * 4);
  float*  dAv   = (float*) carve((size_t)SEQ * NHEADS * 4);
  float*  dAcs  = (float*) carve((size_t)NCHUNK * NHEADS * CHUNKT * 4);
  float*  atot  = (float*) carve((size_t)NCHUNK * NHEADS * 4);
  __bf16* Bbf   = (__bf16*)carve((size_t)SEQ * DSTATE * 2);
  __bf16* Cbf   = (__bf16*)carve((size_t)SEQ * DSTATE * 2);
  __bf16* BbfT  = (__bf16*)carve((size_t)DSTATE * SEQ * 2);
  __bf16* xdtT  = (__bf16*)carve((size_t)DINNER * SEQ * 2);
  __bf16* xddT  = (__bf16*)carve((size_t)DINNER * SEQ * 2);
  float*  st    = (float*) carve((size_t)NCHUNK * NHEADS * HEAD * DSTATE * 4);
  __bf16* pst   = (__bf16*)carve((size_t)NCHUNK * NHEADS * HEAD * DSTATE * 2);
  float*  yb    = (float*) carve((size_t)SEQ * DINNER * 4);
  __bf16* ynbf  = (__bf16*)carve((size_t)SEQ * DINNER * 2);
  __bf16* wbuf  = (__bf16*)carve((size_t)NPAD_IP * DINNER * 2);  // max weight stage
  __bf16* embbf = (__bf16*)carve((size_t)VOCAB * DMODEL * 2);

  {
    long total = (long)VOCAB * DMODEL;
    cvt_w_kernel<<<(int)((total + 255) / 256), 256, 0, stream>>>(
        emb, embbf, VOCAB, VOCAB, DMODEL);
  }

  embed_kernel<<<SEQ, 256, 0, stream>>>(ids, emb, x0);

  float* xcur = x0;
  float* xnext = x1;
  for (int i = 0; i < NLAYER; ++i) {
    rmsnorm_bf16_kernel<<<SEQ, 256, 0, stream>>>(xcur, lnw + (long)i * DMODEL,
                                                 hnbf, DMODEL, 1e-6f);
    {
      long total = (long)NPAD_IP * DMODEL;
      cvt_w_kernel<<<(int)((total + 255) / 256), 256, 0, stream>>>(
          ipw + (long)i * DINPROJ * DMODEL, wbuf, DINPROJ, NPAD_IP, DMODEL);
    }
    gemm_nt_bf16_wmma<<<dim3(SEQ / 64, NPAD_IP / 128), 256, 0, stream>>>(
        hnbf, wbuf, zx, nullptr, SEQ, DINPROJ, DMODEL);
    {
      long total = (long)SEQ * CONVDIM;
      conv_silu_kernel<<<(int)((total + 255) / 256), 256, 0, stream>>>(
          zx, cwv + (long)i * CONVDIM * 4, cbv + (long)i * CONVDIM, xc);
    }
    dt_kernel<<<(SEQ * NHEADS + 255) / 256, 256, 0, stream>>>(
        zx, dtb + i * NHEADS, alog + i * NHEADS, dtv, dAv);
    cumsum_kernel<<<(NCHUNK * NHEADS + 255) / 256, 256, 0, stream>>>(dAv, dAcs, atot);
    stage_bc_kernel<<<(SEQ * DSTATE + 255) / 256, 256, 0, stream>>>(xc, Bbf, Cbf, BbfT);
    {
      long total = (long)SEQ * DINNER;
      stage_x_kernel<<<(int)((total + 255) / 256), 256, 0, stream>>>(
          xc, dtv, dAcs, atot, xdtT, xddT);
    }
    ssd_chunk_kernel<<<dim3(NCHUNK, NHEADS), 256, 0, stream>>>(
        Bbf, Cbf, BbfT, xdtT, xddT, dAcs, yb, st);
    ssd_scan_kernel<<<NHEADS, 256, 0, stream>>>(st, atot, pst);
    ssd_off_kernel<<<dim3(NCHUNK, NHEADS), 256, 0, stream>>>(
        Cbf, pst, dAcs, xc, dsk + i * NHEADS, yb);
    gated_norm_kernel<<<SEQ, 256, 0, stream>>>(yb, zx, gwv + (long)i * DINNER, ynbf);
    {
      long total = (long)DMODEL * DINNER;
      cvt_w_kernel<<<(int)((total + 255) / 256), 256, 0, stream>>>(
          opw + (long)i * DMODEL * DINNER, wbuf, DMODEL, DMODEL, DINNER);
    }
    gemm_nt_bf16_wmma<<<dim3(SEQ / 64, DMODEL / 128), 256, 0, stream>>>(
        ynbf, wbuf, xnext, xcur, SEQ, DMODEL, DINNER);
    float* tmp = xcur; xcur = xnext; xnext = tmp;
  }

  rmsnorm_bf16_kernel<<<SEQ, 256, 0, stream>>>(xcur, nfw, hnbf, DMODEL, 1e-6f);
  gemm_nt_bf16_wmma<<<dim3(SEQ / 64, VOCAB / 128), 256, 0, stream>>>(
      hnbf, embbf, (float*)d_out, nullptr, SEQ, VOCAB, DMODEL);

  (void)in_sizes; (void)n_in; (void)out_size; (void)ws_size;
}
